// DualEncoderEpsNetwork_26869315404289
// MI455X (gfx1250) — matrix operand order repaired and checked by reading, not compiled, not run on hardware
//
#include <hip/hip_runtime.h>

// ---------------- CDNA5 WMMA types ----------------
typedef __bf16 v16bf __attribute__((ext_vector_type(16)));
typedef __bf16 v8bf  __attribute__((ext_vector_type(8)));
typedef float  v8f   __attribute__((ext_vector_type(8)));

union V16U { v16bf v; v8bf h[2]; };

#define NTOT 20000
#define ETOT 320000
#define HD   128

#define DSWAIT() asm volatile("s_wait_dscnt 0x0" ::: "memory")

__device__ __forceinline__ float relu_f(float x) { return fmaxf(x, 0.0f); }
// shifted softplus: log(1+e^x) - log 2, overflow-safe
__device__ __forceinline__ float ssp_f(float x) {
  return fmaxf(x, 0.0f) + log1pf(__expf(-fabsf(x))) - 0.69314718055994531f;
}

__device__ __forceinline__ void zero_acc(v8f* a, int n) {
  for (int i = 0; i < n; ++i)
    for (int j = 0; j < 8; ++j) a[i][j] = 0.0f;
}

// Wave-level GEMM: D(16 x 16*NT) += A(16 x K) * W(K x 16*NT)
// aT: LDS, 16 rows x K bf16, row-major.  wT: LDS, [n][K] bf16 (W transposed).
template <int K, int NT>
__device__ __forceinline__ void wave_gemm(v8f* acc, const __bf16* aT,
                                          const __bf16* wT, int lane) {
  const int col  = lane & 15;
  const int aoff = (lane < 16) ? 0 : 16;
  const int boff = (lane < 16) ? 0 : 16;
  const __bf16* arow = aT + (lane & 15) * K;
#pragma unroll
  for (int ks = 0; ks < K; ks += 32) {
    V16U a;
    a.h[0] = *(const v8bf*)(arow + ks + aoff);       // K = ks+aoff .. +7
    a.h[1] = *(const v8bf*)(arow + ks + aoff + 8);   // K = ks+aoff+8 .. +15
#pragma unroll
    for (int nt = 0; nt < NT; ++nt) {
      v16bf b = *(const v16bf*)(wT + (nt * 16 + col) * K + ks + boff);
      acc[nt] = __builtin_amdgcn_wmma_f32_16x16x32_bf16(
          false, a.v, false, b, (short)0, acc[nt], false, false);
    }
  }
}

// ---------------- small utility kernels ----------------
__global__ void zero4_kernel(float* __restrict__ p, long n) {
  long i = ((long)blockIdx.x * blockDim.x + threadIdx.x) * 4;
  if (i < n) *(float4*)(p + i) = make_float4(0.f, 0.f, 0.f, 0.f);
}

__global__ void embed_kernel(const int* __restrict__ atype,
                             const float* __restrict__ sch_emb,
                             const float* __restrict__ gin_emb,
                             float* __restrict__ hg, float* __restrict__ xl) {
  long i = (long)blockIdx.x * blockDim.x + threadIdx.x;  // N*32 float4 lanes
  if (i >= (long)NTOT * 32) return;
  long node = i >> 5;
  int c = (int)(i & 31) << 2;
  int a = atype[node];
  *(float4*)(hg + node * HD + c) = *(const float4*)(sch_emb + (long)a * HD + c);
  *(float4*)(xl + node * HD + c) = *(const float4*)(gin_emb + (long)a * HD + c);
}

// msg = (h@cW1)[src] * Wfilter ; scatter-add into agg[dst]
__global__ void sch_msg_kernel(const float* __restrict__ hm,
                               const float* __restrict__ wbuf,
                               const int* __restrict__ src,
                               const int* __restrict__ dst,
                               float* __restrict__ agg) {
  long i = (long)blockIdx.x * blockDim.x + threadIdx.x;
  if (i >= (long)ETOT * 32) return;
  long e = i >> 5;
  int c = (int)(i & 31) << 2;
  int s = src[e], d = dst[e];
  float4 h = *(const float4*)(hm + (long)s * HD + c);
  float4 wv = *(const float4*)(wbuf + e * HD + c);
  float* ap = agg + (long)d * HD + c;
  atomicAdd(ap + 0, h.x * wv.x);
  atomicAdd(ap + 1, h.y * wv.y);
  atomicAdd(ap + 2, h.z * wv.z);
  atomicAdd(ap + 3, h.w * wv.w);
}

// msg = relu(x[src]+ea_l)*lm ; scatter-add into agg[dst]
__global__ void gin_msg_kernel(const float* __restrict__ x,
                               const float* __restrict__ ea,
                               const int* __restrict__ etype,
                               const int* __restrict__ src,
                               const int* __restrict__ dst,
                               float* __restrict__ agg) {
  long i = (long)blockIdx.x * blockDim.x + threadIdx.x;
  if (i >= (long)ETOT * 32) return;
  long e = i >> 5;
  if (etype[e] == 0) return;  // local_edge_mask == 0
  int c = (int)(i & 31) << 2;
  int s = src[e], d = dst[e];
  float4 xv = *(const float4*)(x + (long)s * HD + c);
  float4 ev = *(const float4*)(ea + e * HD + c);
  float* ap = agg + (long)d * HD + c;
  atomicAdd(ap + 0, relu_f(xv.x + ev.x));
  atomicAdd(ap + 1, relu_f(xv.y + ev.y));
  atomicAdd(ap + 2, relu_f(xv.z + ev.z));
  atomicAdd(ap + 3, relu_f(xv.w + ev.w));
}

// ---------------- edge encoder: out = (relu(len*W0+b0)@W1 + b1) * bemb[etype] ----------------
__global__ __launch_bounds__(256) void edge_enc_kernel(
    const float* __restrict__ elen, const int* __restrict__ etype,
    const float* __restrict__ W0, const float* __restrict__ b0,
    const float* __restrict__ W1, const float* __restrict__ b1,
    const float* __restrict__ bemb, float* __restrict__ out) {
  __shared__ __align__(64) __bf16 wt[HD * HD];
  __shared__ __align__(64) __bf16 atile[8][16 * HD];
  __shared__ float sW0[HD], sb0[HD], sb1[HD];
  __shared__ float slen[8][16];
  __shared__ int   sety[8][16];
  int tid = threadIdx.x;
  for (int i = tid; i < HD * HD; i += 256) {
    int k = i >> 7, n = i & 127;
    wt[n * HD + k] = (__bf16)W1[i];
  }
  if (tid < HD) { sW0[tid] = W0[tid]; sb0[tid] = b0[tid]; sb1[tid] = b1[tid]; }
  __syncthreads();
  int w = tid >> 5, lane = tid & 31, col = lane & 15, roff = (lane < 16) ? 0 : 8;
  long e0 = (long)blockIdx.x * 128 + w * 16;
  if (lane < 16) { slen[w][lane] = elen[e0 + lane]; sety[w][lane] = etype[e0 + lane]; }
  DSWAIT();
  __bf16* at = &atile[w][0];
  for (int t = lane; t < 16 * 64; t += 32) {
    int m = t >> 6, kp = (t & 63) << 1;
    float l = slen[w][m];
    at[m * HD + kp]     = (__bf16)relu_f(l * sW0[kp] + sb0[kp]);
    at[m * HD + kp + 1] = (__bf16)relu_f(l * sW0[kp + 1] + sb0[kp + 1]);
  }
  DSWAIT();
  v8f acc[8]; zero_acc(acc, 8);
  wave_gemm<HD, 8>(acc, at, wt, lane);
  for (int nt = 0; nt < 8; ++nt) {
    int n = nt * 16 + col;
    float bias = sb1[n];
    for (int r = 0; r < 8; ++r) {
      int m = r + roff;
      float v = (acc[nt][r] + bias) * bemb[(long)sety[w][m] * HD + n];
      out[(e0 + m) * HD + n] = v;
    }
  }
}

// ---------------- SchNet filter: wbuf = (ssp(ea@fW0+fb0)@fW1 + fb1) * (len<=10) ----------------
__global__ __launch_bounds__(256) void filter_kernel(
    const float* __restrict__ ea, const float* __restrict__ elen,
    const float* __restrict__ fW0, const float* __restrict__ fb0,
    const float* __restrict__ fW1, const float* __restrict__ fb1,
    float* __restrict__ wbuf) {
  __shared__ __align__(64) __bf16 wt0[HD * HD];
  __shared__ __align__(64) __bf16 wt1[HD * HD];
  __shared__ __align__(64) __bf16 atile[8][16 * HD];
  __shared__ float sb0[HD], sb1[HD];
  __shared__ float slen[8][16];
  int tid = threadIdx.x;
  for (int i = tid; i < HD * HD; i += 256) {
    int k = i >> 7, n = i & 127;
    wt0[n * HD + k] = (__bf16)fW0[i];
    wt1[n * HD + k] = (__bf16)fW1[i];
  }
  if (tid < HD) { sb0[tid] = fb0[tid]; sb1[tid] = fb1[tid]; }
  __syncthreads();
  int w = tid >> 5, lane = tid & 31, col = lane & 15, roff = (lane < 16) ? 0 : 8;
  long e0 = (long)blockIdx.x * 128 + w * 16;
  if (lane < 16) slen[w][lane] = elen[e0 + lane];
  __bf16* at = &atile[w][0];
  for (int t = lane; t < 16 * 64; t += 32) {
    int m = t >> 6, kp = (t & 63) << 1;
    const float* p = ea + (e0 + m) * HD + kp;
    at[m * HD + kp] = (__bf16)p[0];
    at[m * HD + kp + 1] = (__bf16)p[1];
  }
  DSWAIT();
  v8f acc[8]; zero_acc(acc, 8);
  wave_gemm<HD, 8>(acc, at, wt0, lane);
  for (int nt = 0; nt < 8; ++nt) {
    int n = nt * 16 + col;
    float b = sb0[n];
    for (int r = 0; r < 8; ++r) at[(r + roff) * HD + n] = (__bf16)ssp_f(acc[nt][r] + b);
  }
  DSWAIT();
  v8f acc2[8]; zero_acc(acc2, 8);
  wave_gemm<HD, 8>(acc2, at, wt1, lane);
  for (int nt = 0; nt < 8; ++nt) {
    int n = nt * 16 + col;
    float b = sb1[n];
    for (int r = 0; r < 8; ++r) {
      int m = r + roff;
      float C = (slen[w][m] <= 10.0f) ? 1.0f : 0.0f;
      wbuf[(e0 + m) * HD + n] = (acc2[nt][r] + b) * C;
    }
  }
}

// ---------------- plain node GEMM: out = in @ W ----------------
__global__ __launch_bounds__(256) void gemm_nn_kernel(const float* __restrict__ in,
                                                      const float* __restrict__ W,
                                                      float* __restrict__ out, int rows) {
  __shared__ __align__(64) __bf16 wt[HD * HD];
  __shared__ __align__(64) __bf16 atile[8][16 * HD];
  int tid = threadIdx.x;
  for (int i = tid; i < HD * HD; i += 256) {
    int k = i >> 7, n = i & 127;
    wt[n * HD + k] = (__bf16)W[i];
  }
  __syncthreads();
  int w = tid >> 5, lane = tid & 31, col = lane & 15, roff = (lane < 16) ? 0 : 8;
  long e0 = (long)blockIdx.x * 128 + w * 16;
  if (e0 >= rows) return;
  __bf16* at = &atile[w][0];
  for (int t = lane; t < 16 * 64; t += 32) {
    int m = t >> 6, kp = (t & 63) << 1;
    const float* p = in + (e0 + m) * HD + kp;
    at[m * HD + kp] = (__bf16)p[0];
    at[m * HD + kp + 1] = (__bf16)p[1];
  }
  DSWAIT();
  v8f acc[8]; zero_acc(acc, 8);
  wave_gemm<HD, 8>(acc, at, wt, lane);
  for (int nt = 0; nt < 8; ++nt) {
    int n = nt * 16 + col;
    for (int r = 0; r < 8; ++r) out[(e0 + r + roff) * HD + n] = acc[nt][r];
  }
}

// ---------------- SchNet update: h += ssp(agg@cW2+cb2)@oW + ob ----------------
__global__ __launch_bounds__(256) void sch_update_kernel(
    const float* __restrict__ agg, const float* __restrict__ cW2,
    const float* __restrict__ cb2, const float* __restrict__ oW,
    const float* __restrict__ ob, float* __restrict__ hg, int rows) {
  __shared__ __align__(64) __bf16 wt0[HD * HD];
  __shared__ __align__(64) __bf16 wt1[HD * HD];
  __shared__ __align__(64) __bf16 atile[8][16 * HD];
  __shared__ float sb0[HD], sb1[HD];
  int tid = threadIdx.x;
  for (int i = tid; i < HD * HD; i += 256) {
    int k = i >> 7, n = i & 127;
    wt0[n * HD + k] = (__bf16)cW2[i];
    wt1[n * HD + k] = (__bf16)oW[i];
  }
  if (tid < HD) { sb0[tid] = cb2[tid]; sb1[tid] = ob[tid]; }
  __syncthreads();
  int w = tid >> 5, lane = tid & 31, col = lane & 15, roff = (lane < 16) ? 0 : 8;
  long e0 = (long)blockIdx.x * 128 + w * 16;
  if (e0 >= rows) return;
  __bf16* at = &atile[w][0];
  for (int t = lane; t < 16 * 64; t += 32) {
    int m = t >> 6, kp = (t & 63) << 1;
    const float* p = agg + (e0 + m) * HD + kp;
    at[m * HD + kp] = (__bf16)p[0];
    at[m * HD + kp + 1] = (__bf16)p[1];
  }
  DSWAIT();
  v8f acc[8]; zero_acc(acc, 8);
  wave_gemm<HD, 8>(acc, at, wt0, lane);
  for (int nt = 0; nt < 8; ++nt) {
    int n = nt * 16 + col;
    float b = sb0[n];
    for (int r = 0; r < 8; ++r) at[(r + roff) * HD + n] = (__bf16)ssp_f(acc[nt][r] + b);
  }
  DSWAIT();
  v8f acc2[8]; zero_acc(acc2, 8);
  wave_gemm<HD, 8>(acc2, at, wt1, lane);
  for (int nt = 0; nt < 8; ++nt) {
    int n = nt * 16 + col;
    float b = sb1[n];
    for (int r = 0; r < 8; ++r) {
      long idx = (e0 + r + roff) * HD + n;
      hg[idx] = hg[idx] + acc2[nt][r] + b;
    }
  }
}

// ---------------- GIN update: x = act(relu((x+agg)@W0+b0)@W1+b1) + x ----------------
__global__ __launch_bounds__(256) void gin_update_kernel(
    float* __restrict__ x, const float* __restrict__ agg,
    const float* __restrict__ W0, const float* __restrict__ b0,
    const float* __restrict__ W1, const float* __restrict__ b1,
    int final_relu, int rows) {
  __shared__ __align__(64) __bf16 wt0[HD * HD];
  __shared__ __align__(64) __bf16 wt1[HD * HD];
  __shared__ __align__(64) __bf16 atile[8][16 * HD];
  __shared__ float sb0[HD], sb1[HD];
  int tid = threadIdx.x;
  for (int i = tid; i < HD * HD; i += 256) {
    int k = i >> 7, n = i & 127;
    wt0[n * HD + k] = (__bf16)W0[i];
    wt1[n * HD + k] = (__bf16)W1[i];
  }
  if (tid < HD) { sb0[tid] = b0[tid]; sb1[tid] = b1[tid]; }
  __syncthreads();
  int w = tid >> 5, lane = tid & 31, col = lane & 15, roff = (lane < 16) ? 0 : 8;
  long e0 = (long)blockIdx.x * 128 + w * 16;
  if (e0 >= rows) return;
  __bf16* at = &atile[w][0];
  for (int t = lane; t < 16 * 64; t += 32) {
    int m = t >> 6, kp = (t & 63) << 1;
    long base = (e0 + m) * HD + kp;
    at[m * HD + kp]     = (__bf16)(x[base] + agg[base]);
    at[m * HD + kp + 1] = (__bf16)(x[base + 1] + agg[base + 1]);
  }
  DSWAIT();
  v8f acc[8]; zero_acc(acc, 8);
  wave_gemm<HD, 8>(acc, at, wt0, lane);
  for (int nt = 0; nt < 8; ++nt) {
    int n = nt * 16 + col;
    float b = sb0[n];
    for (int r = 0; r < 8; ++r) at[(r + roff) * HD + n] = (__bf16)relu_f(acc[nt][r] + b);
  }
  DSWAIT();
  v8f acc2[8]; zero_acc(acc2, 8);
  wave_gemm<HD, 8>(acc2, at, wt1, lane);
  for (int nt = 0; nt < 8; ++nt) {
    int n = nt * 16 + col;
    float b = sb1[n];
    for (int r = 0; r < 8; ++r) {
      long idx = (e0 + r + roff) * HD + n;
      float v = acc2[nt][r] + b;
      if (final_relu) v = relu_f(v);
      x[idx] = v + x[idx];  // short_cut
    }
  }
}

// ---------------- pair head: out = relu(relu(cat(n[s]*n[d], ea)@W0+b0)@W1+b1)@W2+b2 (*lm) ----------------
__global__ __launch_bounds__(256) void pair_kernel(
    const float* __restrict__ node, const float* __restrict__ ea,
    const int* __restrict__ src, const int* __restrict__ dst,
    const float* __restrict__ W0, const float* __restrict__ b0,
    const float* __restrict__ W1, const float* __restrict__ b1,
    const float* __restrict__ W2, const float* __restrict__ b2,
    const int* __restrict__ etype, float* __restrict__ out) {
  __shared__ __align__(64) __bf16 wt0[HD * 256];  // [n][256]
  __shared__ __align__(64) __bf16 wt1[64 * HD];   // [n][128]
  __shared__ __align__(64) __bf16 atile[8][16 * 256];
  __shared__ float sb0[HD], sb1[64], sW2[64];
  __shared__ int ssrc[8][16], sdst[8][16], sety[8][16];
  int tid = threadIdx.x;
  for (int i = tid; i < 256 * HD; i += 256) {
    int k = i >> 7, n = i & 127;
    wt0[n * 256 + k] = (__bf16)W0[i];
  }
  for (int i = tid; i < HD * 64; i += 256) {
    int k = i >> 6, n = i & 63;
    wt1[n * HD + k] = (__bf16)W1[i];
  }
  if (tid < HD) sb0[tid] = b0[tid];
  if (tid < 64) { sb1[tid] = b1[tid]; sW2[tid] = W2[tid]; }
  __syncthreads();
  int w = tid >> 5, lane = tid & 31, col = lane & 15, roff = (lane < 16) ? 0 : 8;
  long e0 = (long)blockIdx.x * 128 + w * 16;
  float bb2 = b2[0];
  if (lane < 16) {
    ssrc[w][lane] = src[e0 + lane];
    sdst[w][lane] = dst[e0 + lane];
    sety[w][lane] = etype ? etype[e0 + lane] : 1;
  }
  DSWAIT();
  __bf16* at = &atile[w][0];
  for (int t = lane; t < 16 * 128; t += 32) {
    int m = t >> 7, kp = (t & 127) << 1;
    float v0, v1;
    if (kp < 128) {
      long s = (long)ssrc[w][m] * HD, d = (long)sdst[w][m] * HD;
      v0 = node[s + kp] * node[d + kp];
      v1 = node[s + kp + 1] * node[d + kp + 1];
    } else {
      const float* p = ea + (e0 + m) * HD + (kp - 128);
      v0 = p[0]; v1 = p[1];
    }
    at[m * 256 + kp] = (__bf16)v0;
    at[m * 256 + kp + 1] = (__bf16)v1;
  }
  DSWAIT();
  v8f acc[8]; zero_acc(acc, 8);
  wave_gemm<256, 8>(acc, at, wt0, lane);
  // relu, re-pack as 16x128 tile (stride 128) in-place
  for (int nt = 0; nt < 8; ++nt) {
    int n = nt * 16 + col;
    float b = sb0[n];
    for (int r = 0; r < 8; ++r) at[(r + roff) * HD + n] = (__bf16)relu_f(acc[nt][r] + b);
  }
  DSWAIT();
  v8f acc2[4]; zero_acc(acc2, 4);
  wave_gemm<HD, 4>(acc2, at, wt1, lane);
  // final 64-wide dot with W2, reduced across 16-lane halves
  float pr[8];
  for (int r = 0; r < 8; ++r) pr[r] = 0.0f;
  for (int nt = 0; nt < 4; ++nt) {
    int n = nt * 16 + col;
    float b = sb1[n], w2 = sW2[n];
    for (int r = 0; r < 8; ++r) pr[r] += relu_f(acc2[nt][r] + b) * w2;
  }
  for (int r = 0; r < 8; ++r) {
    float s = pr[r];
    for (int off = 8; off > 0; off >>= 1) s += __shfl_xor(s, off, 16);
    if (col == 0) {
      int m = r + roff;
      float lm = (sety[w][m] != 0) ? 1.0f : 0.0f;
      out[e0 + m] = (s + bb2) * lm;
    }
  }
}

// ---------------- host orchestration ----------------
extern "C" void kernel_launch(void* const* d_in, const int* in_sizes, int n_in,
                              void* d_out, int out_size, void* d_ws, size_t ws_size,
                              hipStream_t stream) {
  const int*   atom_type   = (const int*)d_in[0];
  const int*   edge_index  = (const int*)d_in[1];
  const int*   edge_type   = (const int*)d_in[2];
  const float* edge_length = (const float*)d_in[3];
  const float* eg_bemb = (const float*)d_in[4];
  const float* eg_W0 = (const float*)d_in[5];
  const float* eg_b0 = (const float*)d_in[6];
  const float* eg_W1 = (const float*)d_in[7];
  const float* eg_b1 = (const float*)d_in[8];
  const float* el_bemb = (const float*)d_in[9];
  const float* el_W0 = (const float*)d_in[10];
  const float* el_b0 = (const float*)d_in[11];
  const float* el_W1 = (const float*)d_in[12];
  const float* el_b1 = (const float*)d_in[13];
  const float* sch_emb = (const float*)d_in[14];
  const float* sch_fW0 = (const float*)d_in[15];
  const float* sch_fW1 = (const float*)d_in[16];
  const float* sch_cW1 = (const float*)d_in[17];
  const float* sch_cW2 = (const float*)d_in[18];
  const float* sch_oW  = (const float*)d_in[19];
  const float* sch_fb0 = (const float*)d_in[20];
  const float* sch_fb1 = (const float*)d_in[21];
  const float* sch_cb2 = (const float*)d_in[22];
  const float* sch_ob  = (const float*)d_in[23];
  const float* gin_emb = (const float*)d_in[24];
  const float* gin_W0  = (const float*)d_in[25];
  const float* gin_b0  = (const float*)d_in[26];
  const float* gin_W1  = (const float*)d_in[27];
  const float* gin_b1  = (const float*)d_in[28];
  const float* gg_W0 = (const float*)d_in[29];
  const float* gg_b0 = (const float*)d_in[30];
  const float* gg_W1 = (const float*)d_in[31];
  const float* gg_b1 = (const float*)d_in[32];
  const float* gg_W2 = (const float*)d_in[33];
  const float* gg_b2 = (const float*)d_in[34];
  const float* gl_W0 = (const float*)d_in[35];
  const float* gl_b0 = (const float*)d_in[36];
  const float* gl_W1 = (const float*)d_in[37];
  const float* gl_b1 = (const float*)d_in[38];
  const float* gl_W2 = (const float*)d_in[39];
  const float* gl_b2 = (const float*)d_in[40];

  const int* src = edge_index;
  const int* dst = edge_index + ETOT;

  // workspace layout (f32)
  float* ws = (float*)d_ws;
  const size_t EH = (size_t)ETOT * HD, NH = (size_t)NTOT * HD, HH = (size_t)HD * HD;
  float* ea_g = ws;
  float* ea_l = ea_g + EH;
  float* wbuf = ea_l + EH;
  float* hg   = wbuf + EH;
  float* hm   = hg + NH;
  float* agg  = hm + NH;
  float* xl   = agg + NH;

  const int EDGE_BLKS = ETOT / 128;                 // 2500
  const int NODE_BLKS = (NTOT + 127) / 128;         // 157
  const int MSG_BLKS  = (int)(((long)ETOT * 32) / 256);  // 40000
  const int ZERO_BLKS = (int)((NH / 4 + 255) / 256);

  embed_kernel<<<2500, 256, 0, stream>>>(atom_type, sch_emb, gin_emb, hg, xl);
  edge_enc_kernel<<<EDGE_BLKS, 256, 0, stream>>>(edge_length, edge_type, eg_W0, eg_b0,
                                                 eg_W1, eg_b1, eg_bemb, ea_g);
  edge_enc_kernel<<<EDGE_BLKS, 256, 0, stream>>>(edge_length, edge_type, el_W0, el_b0,
                                                 el_W1, el_b1, el_bemb, ea_l);

  // ---- SchNet global encoder ----
  for (int i = 0; i < 3; ++i) {
    filter_kernel<<<EDGE_BLKS, 256, 0, stream>>>(ea_g, edge_length,
        sch_fW0 + i * HH, sch_fb0 + i * HD, sch_fW1 + i * HH, sch_fb1 + i * HD, wbuf);
    gemm_nn_kernel<<<NODE_BLKS, 256, 0, stream>>>(hg, sch_cW1 + i * HH, hm, NTOT);
    zero4_kernel<<<ZERO_BLKS, 256, 0, stream>>>(agg, (long)NH);
    sch_msg_kernel<<<MSG_BLKS, 256, 0, stream>>>(hm, wbuf, src, dst, agg);
    sch_update_kernel<<<NODE_BLKS, 256, 0, stream>>>(agg, sch_cW2 + i * HH,
        sch_cb2 + i * HD, sch_oW + i * HH, sch_ob + i * HD, hg, NTOT);
  }
  pair_kernel<<<EDGE_BLKS, 256, 0, stream>>>(hg, ea_g, src, dst, gg_W0, gg_b0, gg_W1,
                                             gg_b1, gg_W2, gg_b2, (const int*)nullptr,
                                             (float*)d_out);

  // ---- GIN local encoder ----
  for (int i = 0; i < 3; ++i) {
    zero4_kernel<<<ZERO_BLKS, 256, 0, stream>>>(agg, (long)NH);
    gin_msg_kernel<<<MSG_BLKS, 256, 0, stream>>>(xl, ea_l, edge_type, src, dst, agg);
    gin_update_kernel<<<NODE_BLKS, 256, 0, stream>>>(xl, agg, gin_W0 + i * HH,
        gin_b0 + i * HD, gin_W1 + i * HH, gin_b1 + i * HD, (i < 2) ? 1 : 0, NTOT);
  }
  pair_kernel<<<EDGE_BLKS, 256, 0, stream>>>(xl, ea_l, src, dst, gl_W0, gl_b0, gl_W1,
                                             gl_b1, gl_W2, gl_b2, edge_type,
                                             (float*)d_out + ETOT);
}
